// Model_31095563223360
// MI455X (gfx1250) — matrix operand order repaired and checked by reading, not compiled, not run on hardware
//
#include <hip/hip_runtime.h>

#define BEPS 1e-5f

typedef __attribute__((ext_vector_type(16))) __bf16 v16bf;
typedef __attribute__((ext_vector_type(8)))  float  v8f;
typedef unsigned int u32x4 __attribute__((ext_vector_type(4)));
typedef int          i32x4 __attribute__((ext_vector_type(4)));
typedef int          i32x8 __attribute__((ext_vector_type(8)));

union BfPack { unsigned short us[16]; v16bf v; };

__device__ __forceinline__ unsigned short f2bf(float f) {
  union { float f; unsigned u; } x; x.f = f;
  unsigned r = x.u + 0x7FFFu + ((x.u >> 16) & 1u);
  return (unsigned short)(r >> 16);
}

__device__ __forceinline__ v8f wmma_bf16(v16bf a, v16bf b, v8f c) {
  return __builtin_amdgcn_wmma_f32_16x16x32_bf16(false, a, false, b, (short)0, c, false, false);
}

__device__ __forceinline__ unsigned lds_off(const void* p) {
  return (unsigned)(size_t)p;   // low 32 bits of generic LDS address = LDS byte offset
}

// ---- A fragment: one im2col row (dense 16-bit A 16x32 layout per ISA) ----
// lane: row = lane&15, hi = lane>>4 ; element e -> K = 32*s + (e<8 ? hi*8+e : 16+hi*8+(e-8))
__device__ __forceinline__ v16bf pack_a_row(const unsigned short* row, int s, int hi) {
  BfPack p;
#pragma unroll
  for (int e = 0; e < 16; ++e) {
    int K = 32 * s + (e < 8 ? hi * 8 + e : 16 + hi * 8 + (e - 8));
    p.us[e] = row[K];
  }
  return p.v;
}

// ---- A fragment for conv, channel-major LDS activations, K' = ci*16+kk ----
__device__ __forceinline__ v16bf pack_a_conv(const unsigned short* X, int rs, int s, int p0, int hi) {
  BfPack p;
#pragma unroll
  for (int e = 0; e < 16; ++e) {
    int ci = 2 * s + (e >= 8 ? 1 : 0);
    int kk = hi * 8 + (e & 7);
    p.us[e] = X[ci * rs + p0 + kk];
  }
  return p.v;
}

// ---- B fragment from conv weight (CO,CI,15) f32; K' = ci*16+kk, kk=15 zero slot ----
// B layout (dense 16-bit B 32x16): lane col n = lane&15; element e -> ci=2s+(e>=8), kk=hi*8+(e&7)
__device__ __forceinline__ v16bf pack_b_conv(const float* W, int CI, int co_valid, int n, int s, int hi) {
  BfPack p;
#pragma unroll
  for (int e = 0; e < 16; ++e) {
    int ci = 2 * s + (e >= 8 ? 1 : 0);
    int kk = hi * 8 + (e & 7);
    bool cok = (n < co_valid) && (ci < CI);
    if (kk < 15) {
      int base = cok ? (n * CI + ci) * 15 : 0;
      float v = W[base + (cok ? kk : 0)];
      p.us[e] = cok ? f2bf(v) : (unsigned short)0;
    } else {
      p.us[e] = 0;
    }
  }
  return p.v;
}

// ---- D writebacks, fused BN + ReLU, branchless over tile rows ----
// D element (m,n): per lane n = lane&15, m = hi*8+j
__device__ __forceinline__ void wb_bf16_relu(v8f c, unsigned short* dst, int rs, int n, bool nv,
                                             int t0, int hi, float sc, float bi) {
  if (nv) {
    unsigned* d32 = (unsigned*)(dst + n * rs + t0 + hi * 8);  // even element offset -> 4B aligned
#pragma unroll
    for (int j = 0; j < 4; ++j) {
      float y0 = fmaxf(c[2 * j] * sc + bi, 0.f);
      float y1 = fmaxf(c[2 * j + 1] * sc + bi, 0.f);
      d32[j] = (unsigned)f2bf(y0) | ((unsigned)f2bf(y1) << 16);
    }
  }
}

__device__ __forceinline__ void wb_f32_relu(v8f c, float* dst, int rs, int n, bool nv,
                                            int t0, int hi, float sc, float bi) {
  if (nv) {
    float* d = dst + n * rs + t0 + hi * 8;
#pragma unroll
    for (int j = 0; j < 8; ++j) d[j] = fmaxf(c[j] * sc + bi, 0.f);
  }
}

// =====================================================================
// K0: transpose m_lin_w (370,300) f32 -> wT (300,384) bf16, zero padded
// =====================================================================
__global__ __launch_bounds__(256) void prep_wT_kernel(const float* __restrict__ mw,
                                                      unsigned short* __restrict__ wT) {
  int i = blockIdx.x * 256 + threadIdx.x;
  if (i < 300 * 384) {
    int n = i / 384, k = i % 384;
    wT[i] = (k < 370) ? f2bf(mw[k * 300 + n]) : (unsigned short)0;
  }
}

// =====================================================================
// K1: GNN, one block per drug (1024). Adjacency tile via TDM -> LDS.
// =====================================================================
__global__ __launch_bounds__(256) void gnn_kernel(const int* __restrict__ fpt,
                                                  const float* __restrict__ adj,
                                                  const float* __restrict__ emb,
                                                  const float* __restrict__ wg,
                                                  const float* __restrict__ bg,
                                                  float* __restrict__ compound) {
  __shared__ float xs[64 * 50];
  __shared__ float hs[64 * 50];
  __shared__ float Am[64 * 64];
  const int tid = threadIdx.x;
  const int drug = blockIdx.x;

  // --- Tensor Data Mover: 64x64 f32 adjacency tile, global -> LDS (wave 0 issues) ---
  if (tid < 32) {
    unsigned long long ga = (unsigned long long)(size_t)(adj + (size_t)drug * 4096);
    u32x4 g0;
    g0[0] = 1u;                                             // count=1 valid descriptor
    g0[1] = lds_off(Am);                                    // lds_addr
    g0[2] = (unsigned)(ga & 0xFFFFFFFFu);                   // global_addr[31:0]
    g0[3] = (unsigned)((ga >> 32) & 0x1FFFFFFu) | (2u << 30); // addr[56:32] | type=2
    i32x8 g1;
    g1[0] = (int)(2u << 16);      // data_size=4B
    g1[1] = (int)(64u << 16);     // tensor_dim0 = 64 (bits 79:48)
    g1[2] = (int)(64u << 16);     // tensor_dim1 = 64 (bits 111:80)
    g1[3] = (int)(64u << 16);     // tile_dim0 = 64 (bits 127:112)
    g1[4] = 64;                   // tile_dim1 = 64, tile_dim2 = 0
    g1[5] = 64;                   // tensor_dim0_stride = 64
    g1[6] = (int)(4096u << 16);   // tensor_dim1_stride = 4096 (bits 255:208)
    g1[7] = 0;
    i32x4 g2 = {0, 0, 0, 0};
    i32x4 g3 = {0, 0, 0, 0};
    asm volatile("tensor_load_to_lds %0, %1, %2, %3"
                 :: "s"(g0), "s"(g1), "s"(g2), "s"(g3) : "memory");
    __builtin_amdgcn_s_wait_tensorcnt(0);
  }

  for (int i = tid; i < 64 * 50; i += 256) {
    int a = i / 50, dd = i % 50;
    int fp = fpt[drug * 64 + a];
    xs[i] = emb[(size_t)fp * 50 + dd];
  }
  __syncthreads();

  for (int l = 0; l < 3; ++l) {
    const float* W = wg + l * 2500;
    const float* bb = bg + l * 50;
    for (int i = tid; i < 64 * 50; i += 256) {
      int a = i / 50, dd = i % 50;
      float acc = bb[dd];
      for (int k = 0; k < 50; ++k) acc += xs[a * 50 + k] * W[k * 50 + dd];
      hs[i] = fmaxf(acc, 0.f);
    }
    __syncthreads();
    for (int i = tid; i < 64 * 50; i += 256) {
      int a = i / 50, dd = i % 50;
      float acc = 0.f;
      for (int j = 0; j < 64; ++j) acc += Am[a * 64 + j] * hs[j * 50 + dd];
      xs[i] += acc;
    }
    __syncthreads();
  }
  for (int i = tid; i < 50; i += 256) {
    float acc = 0.f;
    for (int a = 0; a < 64; ++a) acc += xs[a * 50 + i];
    compound[(size_t)drug * 50 + i] = acc * (1.f / 64.f);
  }
}

// =====================================================================
// K2: gene CNN branch, one block (8 waves) per sample, pipeline in LDS.
// =====================================================================
__global__ __launch_bounds__(256) void gene_kernel(
    const float* __restrict__ rma, const float* __restrict__ var,
    const int* __restrict__ drug_id, const float* __restrict__ fuse,
    const float* c1w, const float* c1b, const float* b1g, const float* b1b, const float* b1m, const float* b1v,
    const float* c2w, const float* c2b, const float* b2g, const float* b2b, const float* b2m, const float* b2v,
    const float* c3w, const float* c3b, const float* b3g, const float* b3b, const float* b3m, const float* b3v,
    const float* glw, const float* glb, unsigned short* __restrict__ out_gene) {
  __shared__ unsigned short z_bf[1536];        // 1478 valid, zero pad
  __shared__ unsigned short x1[20 * 752];      // conv1 out, ch-major
  __shared__ unsigned short x2[20 * 368];      // conv2 out
  __shared__ float x3[10 * 176];               // conv3 out (f32)
  __shared__ float pooled[10 * 57];

  const int tid = threadIdx.x;
  const int b = blockIdx.x;

  __builtin_prefetch(c2w, 0, 1);
  __builtin_prefetch(c3w, 0, 1);

  // zero everything the WMMAs may touch (pad determinism, no NaN through zero weights)
  for (int i = tid; i < 1536; i += 256) z_bf[i] = 0;
  for (int i = tid; i < 20 * 752; i += 256) x1[i] = 0;
  for (int i = tid; i < 20 * 368; i += 256) x2[i] = 0;

  // z = rma + clip(fuse[drug],0,1) * var
  const int drug = drug_id[b];
  const float* fz = fuse + (size_t)drug * 1478;
  for (int i = tid; i < 1478; i += 256) {
    float fw = fminf(fmaxf(fz[i], 0.f), 1.f);
    float z = rma[(size_t)b * 1478 + i] + fw * var[(size_t)b * 1478 + i];
    z_bf[i] = f2bf(z);
  }
  __syncthreads();

  const int w = tid >> 5, lane = tid & 31, r = lane & 15, hi = lane >> 4;

  // ---- conv1: 1->20, 1478 -> 732 ----
  {
    int nt = w >> 2;
    int n = nt * 16 + r;
    bool nv = n < 20;
    int nn = nv ? n : 0;
    v16bf bw = pack_b_conv(c1w, 1, 20, n, 0, hi);
    float sc = b1g[nn] * rsqrtf(b1v[nn] + BEPS);
    float bi = (c1b[nn] - b1m[nn]) * sc + b1b[nn];
    for (int mt = (w & 3); mt < 46; mt += 4) {
      int t = mt * 16 + r;
      v8f acc = {};
      acc = wmma_bf16(pack_a_row(z_bf + 2 * t, 0, hi), bw, acc);
      wb_bf16_relu(acc, x1, 752, n, nv, mt * 16, hi, sc, bi);
    }
  }
  __syncthreads();

  // ---- conv2: 20->20, 732 -> 359, K=320 ----
  {
    int nt = w >> 2;
    int n = nt * 16 + r;
    bool nv = n < 20;
    int nn = nv ? n : 0;
    v16bf bw[10];
#pragma unroll
    for (int s = 0; s < 10; ++s) bw[s] = pack_b_conv(c2w, 20, 20, n, s, hi);
    float sc = b2g[nn] * rsqrtf(b2v[nn] + BEPS);
    float bi = (c2b[nn] - b2m[nn]) * sc + b2b[nn];
    for (int mt = (w & 3); mt < 23; mt += 4) {
      int t = mt * 16 + r;
      v8f acc = {};
#pragma unroll
      for (int s = 0; s < 10; ++s)
        acc = wmma_bf16(pack_a_conv(x1, 752, s, 2 * t, hi), bw[s], acc);
      wb_bf16_relu(acc, x2, 368, n, nv, mt * 16, hi, sc, bi);
    }
  }
  __syncthreads();

  // ---- conv3: 20->10, 359 -> 173, K=320 ----
  {
    int n = r;
    bool nv = n < 10;
    int nn = nv ? n : 0;
    v16bf bw[10];
#pragma unroll
    for (int s = 0; s < 10; ++s) bw[s] = pack_b_conv(c3w, 20, 10, n, s, hi);
    float sc = b3g[nn] * rsqrtf(b3v[nn] + BEPS);
    float bi = (c3b[nn] - b3m[nn]) * sc + b3b[nn];
    for (int mt = w; mt < 11; mt += 8) {
      int t = mt * 16 + r;
      v8f acc = {};
#pragma unroll
      for (int s = 0; s < 10; ++s)
        acc = wmma_bf16(pack_a_conv(x2, 368, s, 2 * t, hi), bw[s], acc);
      wb_f32_relu(acc, x3, 176, n, nv, mt * 16, hi, sc, bi);
    }
  }
  __syncthreads();

  // ---- maxpool k3 s3 -> 57 ----
  for (int i = tid; i < 570; i += 256) {
    int ch = i / 57, p = i % 57;
    const float* xr = x3 + ch * 176 + 3 * p;
    pooled[i] = fmaxf(fmaxf(xr[0], xr[1]), xr[2]);
  }
  __syncthreads();

  // ---- g_lin (57->32 per channel) + relu, store bf16 ----
  for (int i = tid; i < 320; i += 256) {
    int ch = i >> 5, j = i & 31;
    float acc = glb[j];
    for (int p = 0; p < 57; ++p) acc += pooled[ch * 57 + p] * glw[p * 32 + j];
    out_gene[(size_t)b * 320 + i] = f2bf(fmaxf(acc, 0.f));
  }
}

// =====================================================================
// K3: fusion tail, 16 samples per block. Gene tile via async->LDS,
// m_lin via WMMA, small convs in VALU.
// =====================================================================
__global__ __launch_bounds__(256) void tail_kernel(
    const unsigned short* __restrict__ out_gene, const float* __restrict__ compound,
    const int* __restrict__ drug_id, const unsigned short* __restrict__ wT,
    const float* mlb,
    const float* c1w, const float* c1b, const float* n1g, const float* n1b, const float* n1m, const float* n1v,
    const float* c2w, const float* c2b, const float* n2g, const float* n2b, const float* n2m, const float* n2v,
    const float* ow, const float* ob, float* __restrict__ out) {
  __shared__ unsigned short C[16 * 384];   // concat tile (bf16)
  __shared__ float M[16 * 304];            // tanh(m_lin)
  __shared__ float P1[16 * 10 * 29];
  __shared__ float P2[16 * 10];

  const int tid = threadIdx.x;
  const int b0 = blockIdx.x * 16;

  __builtin_prefetch(wT, 0, 1);

  // gene features: 16 rows x 640B, contiguous -> async global->LDS (b128 per lane)
  for (int i = tid; i < 16 * 40; i += 256) {
    int r = i / 40, c8 = (i % 40) * 8;
    unsigned dst = lds_off(&C[r * 384 + c8]);
    const unsigned short* src = out_gene + (size_t)(b0 + r) * 320 + c8;
    asm volatile("global_load_async_to_lds_b128 %0, %1, off"
                 :: "v"(dst), "v"(src) : "memory");
  }
  // compound + zero pad columns via normal stores
  for (int i = tid; i < 16 * 64; i += 256) {
    int r = i >> 6, c = 320 + (i & 63);
    unsigned short v = 0;
    if (c < 370) v = f2bf(compound[(size_t)drug_id[b0 + r] * 50 + (c - 320)]);
    C[r * 384 + c] = v;
  }
  asm volatile("s_wait_asynccnt 0x0" ::: "memory");
  __syncthreads();

  const int w = tid >> 5, lane = tid & 31, r = lane & 15, hi = lane >> 4;

  // m_lin: (16 x 384) @ (384 x 300), 19 col tiles, 12 K chunks
  for (int nt = w; nt < 19; nt += 8) {
    int n = nt * 16 + r;
    bool nv = n < 300;
    int wbase = (nv ? n : 0) * 384;
    v16bf bw[12];
#pragma unroll
    for (int s = 0; s < 12; ++s) {
      BfPack p;
#pragma unroll
      for (int e = 0; e < 16; ++e) {
        unsigned short v = wT[wbase + 32 * s + (e < 8 ? hi * 8 + e : 16 + hi * 8 + (e - 8))];
        p.us[e] = nv ? v : (unsigned short)0;
      }
      bw[s] = p.v;
    }
    v8f acc = {};
#pragma unroll
    for (int s = 0; s < 12; ++s)
      acc = wmma_bf16(pack_a_row(C + r * 384, s, hi), bw[s], acc);
    float bias = mlb[nv ? n : 0];
#pragma unroll
    for (int j = 0; j < 8; ++j)
      M[(hi * 8 + j) * 304 + n] = tanhf(acc[j] + bias);   // n<=303, pad cols never read
  }
  __syncthreads();

  // m_conv1 (1->10, k10 s2 -> 146) + bn + maxpool(5,5) -> 29
  for (int i = tid; i < 16 * 290; i += 256) {
    int rr = i / 290, rem = i % 290, ch = rem / 29, p = rem % 29;
    float s = n1g[ch] * rsqrtf(n1v[ch] + BEPS);
    float bi = (c1b[ch] - n1m[ch]) * s + n1b[ch];
    float best = -3.4e38f;
    for (int q = 0; q < 5; ++q) {
      int u = 5 * p + q;
      float a = 0.f;
      for (int kk = 0; kk < 10; ++kk) a += M[rr * 304 + 2 * u + kk] * c1w[ch * 10 + kk];
      best = fmaxf(best, a * s + bi);
    }
    P1[i] = best;
  }
  __syncthreads();

  // m_conv2 (10->5, k10 s2 -> 10) + bn + maxpool(5,5) -> 2 ; flat ch*2+p
  for (int i = tid; i < 16 * 10; i += 256) {
    int rr = i / 10, j = i % 10, ch = j >> 1, p = j & 1;
    float s = n2g[ch] * rsqrtf(n2v[ch] + BEPS);
    float bi = (c2b[ch] - n2m[ch]) * s + n2b[ch];
    float best = -3.4e38f;
    for (int q = 0; q < 5; ++q) {
      int u = 5 * p + q;
      float a = 0.f;
      for (int ci = 0; ci < 10; ++ci)
        for (int kk = 0; kk < 10; ++kk)
          a += P1[rr * 290 + ci * 29 + 2 * u + kk] * c2w[(ch * 10 + ci) * 10 + kk];
      best = fmaxf(best, a * s + bi);
    }
    P2[i] = best;
  }
  __syncthreads();

  if (tid < 16) {
    float acc = ob[0];
    for (int i = 0; i < 10; ++i) acc += P2[tid * 10 + i] * ow[i];
    out[b0 + tid] = acc;
  }
}

// =====================================================================
extern "C" void kernel_launch(void* const* d_in, const int* in_sizes, int n_in,
                              void* d_out, int out_size, void* d_ws, size_t ws_size,
                              hipStream_t stream) {
  const float* rma  = (const float*)d_in[0];
  const float* var  = (const float*)d_in[1];
  const int*   did  = (const int*)d_in[2];
  const int*   fpt  = (const int*)d_in[3];
  const float* adj  = (const float*)d_in[4];
  const float* fuse = (const float*)d_in[5];
  const float* emb  = (const float*)d_in[6];
  const float* wg   = (const float*)d_in[7];
  const float* bg   = (const float*)d_in[8];
  const float* c1w  = (const float*)d_in[9];   const float* c1b = (const float*)d_in[10];
  const float* b1g  = (const float*)d_in[11];  const float* b1b = (const float*)d_in[12];
  const float* b1m  = (const float*)d_in[13];  const float* b1v = (const float*)d_in[14];
  const float* c2w  = (const float*)d_in[15];  const float* c2b = (const float*)d_in[16];
  const float* b2g  = (const float*)d_in[17];  const float* b2b = (const float*)d_in[18];
  const float* b2m  = (const float*)d_in[19];  const float* b2v = (const float*)d_in[20];
  const float* c3w  = (const float*)d_in[21];  const float* c3b = (const float*)d_in[22];
  const float* b3g  = (const float*)d_in[23];  const float* b3b = (const float*)d_in[24];
  const float* b3m  = (const float*)d_in[25];  const float* b3v = (const float*)d_in[26];
  const float* glw  = (const float*)d_in[27];  const float* glb = (const float*)d_in[28];
  const float* mlw  = (const float*)d_in[29];  const float* mlb = (const float*)d_in[30];
  const float* mc1w = (const float*)d_in[31];  const float* mc1b = (const float*)d_in[32];
  const float* mn1g = (const float*)d_in[33];  const float* mn1b = (const float*)d_in[34];
  const float* mn1m = (const float*)d_in[35];  const float* mn1v = (const float*)d_in[36];
  const float* mc2w = (const float*)d_in[37];  const float* mc2b = (const float*)d_in[38];
  const float* mn2g = (const float*)d_in[39];  const float* mn2b = (const float*)d_in[40];
  const float* mn2m = (const float*)d_in[41];  const float* mn2v = (const float*)d_in[42];
  const float* ow   = (const float*)d_in[43];  const float* ob  = (const float*)d_in[44];
  float* out = (float*)d_out;

  float* compound = (float*)d_ws;                               // 1024*50 f32
  unsigned short* wT = (unsigned short*)(compound + 1024 * 50); // 300*384 bf16
  unsigned short* out_gene = wT + 300 * 384;                    // 4096*320 bf16

  prep_wT_kernel<<<(300 * 384 + 255) / 256, 256, 0, stream>>>(mlw, wT);
  gnn_kernel<<<1024, 256, 0, stream>>>(fpt, adj, emb, wg, bg, compound);
  gene_kernel<<<4096, 256, 0, stream>>>(rma, var, did, fuse,
                                        c1w, c1b, b1g, b1b, b1m, b1v,
                                        c2w, c2b, b2g, b2b, b2m, b2v,
                                        c3w, c3b, b3g, b3b, b3m, b3v,
                                        glw, glb, out_gene);
  tail_kernel<<<4096 / 16, 256, 0, stream>>>(out_gene, compound, did, wT, mlb,
                                             mc1w, mc1b, mn1g, mn1b, mn1m, mn1v,
                                             mc2w, mc2b, mn2g, mn2b, mn2m, mn2v,
                                             ow, ob, out);
  (void)in_sizes; (void)n_in; (void)out_size; (void)ws_size;
}
